// MOFNet_13073880449798
// MI455X (gfx1250) — matrix operand order repaired and checked by reading, not compiled, not run on hardware
//
#include <hip/hip_runtime.h>
#include <hip/hip_bf16.h>

// ---------------------------------------------------------------------------
// MOFNet forward for MI455X (gfx1250, wave32, WMMA).
//
// Heavy work = three (2048x128)@(128x49152) GEMMs fused with tanh and a
// per-sample contraction. Strategy:
//   * convert/transpose weights to fp16 [m][k] once (L2-resident, 37.7MB)
//   * v_wmma_f32_16x16x32_f16 tiles, 16 samples x 16 cols x K=128
//   * tanh via gfx1250 v_tanh_f32, contraction weights staged in LDS
// ---------------------------------------------------------------------------

#define E     128
#define BATCH 2048
#define FIRSTD 768
#define HID1  64
#define HID2  32
#define BN_EPS 1e-3f

typedef __attribute__((ext_vector_type(16))) _Float16 v16h;
typedef __attribute__((ext_vector_type(8)))  _Float16 v8h;
typedef __attribute__((ext_vector_type(8)))  float    v8f;

__device__ __forceinline__ float fast_tanh(float x) {
#if __has_builtin(__builtin_amdgcn_tanhf)
  return __builtin_amdgcn_tanhf(x);            // v_tanh_f32 on gfx1250
#elif __has_builtin(__builtin_amdgcn_tanh_f32)
  return __builtin_amdgcn_tanh_f32(x);
#else
  float e2 = __expf(2.0f * x);                 // v_exp_f32 fallback
  return 1.0f - 2.0f / (e2 + 1.0f);
#endif
}

// A-fragment for V_WMMA_F32_16X16X32_F16, 16-bit A 16x32 in this code's own k order (not the ISA table order):
// lane L (n=L%16,h=L/16) holds row M=n, K runs [16h..16h+15] (contiguous, matches B).
__device__ __forceinline__ v16h load_afrag(const _Float16* __restrict__ row,
                                           int t, int hh) {
  v8h lo = *(const v8h*)(row + 32 * t + 16 * hh);
  v8h hi = *(const v8h*)(row + 32 * t + 16 * hh + 8);
  v16h r;
#pragma unroll
  for (int e = 0; e < 8; ++e) { r[e] = lo[e]; r[8 + e] = hi[e]; }
  return r;
}

// ---------------------------------------------------------------------------
// Prep: transpose W[k=128][M] fp32 -> Wt[M][128] fp16 (LDS-tiled transpose)
// ---------------------------------------------------------------------------
__global__ void transpose_to_f16(const float* __restrict__ W,
                                 _Float16* __restrict__ Wt, int M) {
  __shared__ float s[16][17];
  const int tx = threadIdx.x, ty = threadIdx.y;
  const int m0 = blockIdx.x * 16, i0 = blockIdx.y * 16;
  s[ty][tx] = W[(size_t)(i0 + ty) * M + (m0 + tx)];
  __syncthreads();
  Wt[(size_t)(m0 + ty) * E + (i0 + tx)] = (_Float16)s[tx][ty];
}

// Gather topo embedding rows to fp16 (A operand source, row-major [b][k]).
__global__ void gather_topoA(const int* __restrict__ x,
                             const float* __restrict__ topo_table,
                             _Float16* __restrict__ topoA) {
  int idx = blockIdx.x * blockDim.x + threadIdx.x;   // BATCH*E threads
  int b = idx >> 7, i = idx & 127;
  int t = x[b * 7];
  topoA[idx] = (_Float16)topo_table[(size_t)t * E + i];
}

// ---------------------------------------------------------------------------
// Branch kernel (node or edge):
//   e[b,j,k] = relu(bn( sum_l tanh(topo_emb[b] . Wcol[j,k,l] + bias) * emb[b,j,l] )) * mask
// Block = (sample group of 16, j). 8 waves; wave w owns k = w, w+8, ...
// Per k: 32 WMMAs (16x128 tile, K=128) + tanh + LDS-weighted reduction.
// ---------------------------------------------------------------------------
__launch_bounds__(256, 1)
__global__ void branch_kernel(const _Float16* __restrict__ topoA,
                              const _Float16* __restrict__ Wt,
                              const float* __restrict__ wbias,
                              const float* __restrict__ table,
                              const int* __restrict__ x, int xoff,
                              const float* __restrict__ bn_g,
                              const float* __restrict__ bn_b,
                              const float* __restrict__ bn_m,
                              const float* __restrict__ bn_v,
                              float* __restrict__ cat, int cat_off) {
  __shared__ float embS[16][E];   // 8 KB
  __shared__ float maskS[16];
  const int tid  = threadIdx.x;
  const int g    = blockIdx.x, j = blockIdx.y;
  const int lane = tid & 31, wave = tid >> 5;
  const int n    = lane & 15, hh = lane >> 4;

  // Stage embeddings (with clamp) + row masks for this (group, j).
  for (int idx = tid; idx < 16 * E; idx += 256) {
    int m = idx >> 7, i = idx & 127;
    int b = g * 16 + m;
    int raw = x[b * 7 + xoff + j];
    int c = raw >= 0 ? raw : 0;
    embS[m][i] = table[(size_t)c * E + i];
  }
  if (tid < 16) {
    int b = g * 16 + tid;
    maskS[tid] = (x[b * 7 + xoff + j] >= 0) ? 1.0f : 0.0f;
  }
  __syncthreads();

  // A fragments (same 16 samples for every k/j): 4 k-steps of 32.
  const _Float16* arow = topoA + (size_t)(g * 16 + n) * E;
  v16h a[4];
#pragma unroll
  for (int t = 0; t < 4; ++t) a[t] = load_afrag(arow, t, hh);

  for (int k = wave; k < E; k += 8) {
    v8f acc[8] = {};
    // column m = j*E*E + k*E + l ; Wt row-major [m][128] fp16
    const _Float16* wbase = Wt + ((size_t)(j * E * E + k * E) << 7);
#pragma unroll
    for (int t = 0; t < 4; ++t) {
#pragma unroll
      for (int lt = 0; lt < 8; ++lt) {
        // 16-bit B 32x16 layout: lane holds col N=n, 16 contiguous K = 16*hh..
        const _Float16* brow = wbase + ((size_t)(lt * 16 + n) << 7);
        v16h bf = *(const v16h*)(brow + 32 * t + 16 * hh);
        acc[lt] = __builtin_amdgcn_wmma_f32_16x16x32_f16(
            false, a[t], false, bf, (short)0, acc[lt], false, false);
      }
    }
    // tanh + weighted reduction over l (columns). Lane holds col l=lt*16+n,
    // rows M = 8*hh + r.
    float p[8];
#pragma unroll
    for (int r = 0; r < 8; ++r) p[r] = 0.0f;
    const int mbase = j * E * E + k * E;
#pragma unroll
    for (int lt = 0; lt < 8; ++lt) {
      const int l = lt * 16 + n;
      float bv = wbias[mbase + l];
#pragma unroll
      for (int r = 0; r < 8; ++r) {
        float tt = fast_tanh(acc[lt][r] + bv);
        p[r] = fmaf(tt, embS[8 * hh + r][l], p[r]);
      }
    }
    // Butterfly-sum over the 16 lanes of each half (l dimension).
#pragma unroll
    for (int mask = 1; mask <= 8; mask <<= 1) {
#pragma unroll
      for (int r = 0; r < 8; ++r) p[r] += __shfl_xor(p[r], mask, 32);
    }
    // BN + ReLU + mask, write into cat buffer. Lane n<8 writes row 8*hh+n.
    if (n < 8) {
      int m = 8 * hh + n;
      float ev = p[n];
      float y = bn_g[k] * (ev - bn_m[k]) * rsqrtf(bn_v[k] + BN_EPS) + bn_b[k];
      y = fmaxf(y, 0.0f) * maskS[m];
      cat[(size_t)(g * 16 + m) * FIRSTD + cat_off + j * E + k] = y;
    }
  }
}

// ---------------------------------------------------------------------------
// Inter + tail:
//   h[b,kk] = sum_j tanh(topo_emb[b] . WI[:, j*64+kk] + ib) * cat[b,j]
//   then BN/ReLU -> 64x32 -> BN/ReLU -> 32x1 + bias.
// Block = sample group of 16; 8 waves stride over j (96 each).
// ---------------------------------------------------------------------------
__launch_bounds__(256, 1)
__global__ void inter_kernel(const _Float16* __restrict__ topoA,
                             const _Float16* __restrict__ WtI,
                             const float* __restrict__ ibias,
                             const float* __restrict__ cat,
                             const float* __restrict__ ibn_g,
                             const float* __restrict__ ibn_b,
                             const float* __restrict__ ibn_m,
                             const float* __restrict__ ibn_v,
                             const float* __restrict__ hk,
                             const float* __restrict__ hbn_g,
                             const float* __restrict__ hbn_b,
                             const float* __restrict__ hbn_m,
                             const float* __restrict__ hbn_v,
                             const float* __restrict__ ok,
                             const float* __restrict__ ob,
                             float* __restrict__ out) {
  __shared__ char smem[16 * FIRSTD * 4];   // 48 KB: catS, later reused as hred/hS
  float* catS = (float*)smem;              // [16][768]
  const int tid  = threadIdx.x;
  const int g    = blockIdx.x;
  const int lane = tid & 31, wave = tid >> 5;
  const int n    = lane & 15, hh = lane >> 4;

  for (int idx = tid; idx < 16 * FIRSTD; idx += 256)
    catS[idx] = cat[(size_t)g * 16 * FIRSTD + idx];
  __syncthreads();

  const _Float16* arow = topoA + (size_t)(g * 16 + n) * E;
  v16h a[4];
#pragma unroll
  for (int t = 0; t < 4; ++t) a[t] = load_afrag(arow, t, hh);

  v8f hacc[4] = {};                        // 16 samples x 64 kk
  for (int j = wave; j < FIRSTD; j += 8) {
    v8f tacc[4] = {};
    const _Float16* wbase = WtI + ((size_t)(j * HID1) << 7); // m = j*64 + kk
#pragma unroll
    for (int t = 0; t < 4; ++t) {
#pragma unroll
      for (int nt = 0; nt < 4; ++nt) {
        const _Float16* brow = wbase + ((size_t)(nt * 16 + n) << 7);
        v16h bf = *(const v16h*)(brow + 32 * t + 16 * hh);
        tacc[nt] = __builtin_amdgcn_wmma_f32_16x16x32_f16(
            false, a[t], false, bf, (short)0, tacc[nt], false, false);
      }
    }
    float cv[8];
#pragma unroll
    for (int r = 0; r < 8; ++r) cv[r] = catS[(8 * hh + r) * FIRSTD + j];
#pragma unroll
    for (int nt = 0; nt < 4; ++nt) {
      float bv = ibias[j * HID1 + nt * 16 + n];
#pragma unroll
      for (int r = 0; r < 8; ++r) {
        float tt = fast_tanh(tacc[nt][r] + bv);
        hacc[nt][r] = fmaf(tt, cv[r], hacc[nt][r]);
      }
    }
  }
  __syncthreads();                          // catS no longer needed
  float* hred = (float*)smem;               // [8][16][64] = 32 KB
  float* hS   = (float*)(smem + 32768);     // [16][64]    =  4 KB
#pragma unroll
  for (int nt = 0; nt < 4; ++nt)
#pragma unroll
    for (int r = 0; r < 8; ++r)
      hred[(wave * 16 + 8 * hh + r) * HID1 + nt * 16 + n] = hacc[nt][r];
  __syncthreads();
  for (int idx = tid; idx < 16 * HID1; idx += 256) {
    int m = idx >> 6, kk = idx & 63;
    float s = 0.0f;
#pragma unroll
    for (int w = 0; w < 8; ++w) s += hred[(w * 16 + m) * HID1 + kk];
    float y = ibn_g[kk] * (s - ibn_m[kk]) * rsqrtf(ibn_v[kk] + BN_EPS) + ibn_b[kk];
    hS[m * HID1 + kk] = fmaxf(y, 0.0f);
  }
  __syncthreads();
  if (tid < 16) {                           // one lane per sample: tiny tail
    int b = g * 16 + tid;
    float o = ob[0];
    for (int oo = 0; oo < HID2; ++oo) {
      float s = 0.0f;
#pragma unroll
      for (int kk = 0; kk < HID1; ++kk)
        s = fmaf(hS[tid * HID1 + kk], hk[kk * HID2 + oo], s);
      float y = hbn_g[oo] * (s - hbn_m[oo]) * rsqrtf(hbn_v[oo] + BN_EPS) + hbn_b[oo];
      y = fmaxf(y, 0.0f);
      o = fmaf(y, ok[oo], o);
    }
    out[b] = o;
  }
}

// ---------------------------------------------------------------------------
extern "C" void kernel_launch(void* const* d_in, const int* in_sizes, int n_in,
                              void* d_out, int out_size, void* d_ws, size_t ws_size,
                              hipStream_t stream) {
  const int*   x          = (const int*)  d_in[0];
  const float* topo_table = (const float*)d_in[1];
  const float* node_table = (const float*)d_in[2];
  const float* edge_table = (const float*)d_in[3];
  const float* node_wk    = (const float*)d_in[4];
  const float* node_wb    = (const float*)d_in[5];
  const float* edge_wk    = (const float*)d_in[6];
  const float* edge_wb    = (const float*)d_in[7];
  const float* inter_wk   = (const float*)d_in[8];
  const float* inter_wb   = (const float*)d_in[9];
  const float* nbn_g = (const float*)d_in[10];
  const float* nbn_b = (const float*)d_in[11];
  const float* nbn_m = (const float*)d_in[12];
  const float* nbn_v = (const float*)d_in[13];
  const float* ebn_g = (const float*)d_in[14];
  const float* ebn_b = (const float*)d_in[15];
  const float* ebn_m = (const float*)d_in[16];
  const float* ebn_v = (const float*)d_in[17];
  const float* ibn_g = (const float*)d_in[18];
  const float* ibn_b = (const float*)d_in[19];
  const float* ibn_m = (const float*)d_in[20];
  const float* ibn_v = (const float*)d_in[21];
  const float* hk    = (const float*)d_in[22];
  const float* hbn_g = (const float*)d_in[23];
  const float* hbn_b = (const float*)d_in[24];
  const float* hbn_m = (const float*)d_in[25];
  const float* hbn_v = (const float*)d_in[26];
  const float* ok    = (const float*)d_in[27];
  const float* ob    = (const float*)d_in[28];

  // Workspace layout (~44.6 MB total)
  const size_t WT_BYTES = (size_t)49152 * E * sizeof(_Float16);  // 12,582,912
  char* ws = (char*)d_ws;
  _Float16* WtN   = (_Float16*)(ws);
  _Float16* WtE   = (_Float16*)(ws + WT_BYTES);
  _Float16* WtI   = (_Float16*)(ws + 2 * WT_BYTES);
  _Float16* topoA = (_Float16*)(ws + 3 * WT_BYTES);              // 512 KB
  float*    cat   = (float*)   (ws + 3 * WT_BYTES + (size_t)BATCH * E * 2);

  const int M = 49152;
  dim3 tb(16, 16);
  transpose_to_f16<<<dim3(M / 16, E / 16), tb, 0, stream>>>(node_wk, WtN, M);
  transpose_to_f16<<<dim3(M / 16, E / 16), tb, 0, stream>>>(edge_wk, WtE, M);
  transpose_to_f16<<<dim3(M / 16, E / 16), tb, 0, stream>>>(inter_wk, WtI, M);
  gather_topoA<<<(BATCH * E) / 256, 256, 0, stream>>>(x, topo_table, topoA);

  branch_kernel<<<dim3(BATCH / 16, 3), 256, 0, stream>>>(
      topoA, WtN, node_wb, node_table, x, /*xoff=*/1,
      nbn_g, nbn_b, nbn_m, nbn_v, cat, /*cat_off=*/0);
  branch_kernel<<<dim3(BATCH / 16, 3), 256, 0, stream>>>(
      topoA, WtE, edge_wb, edge_table, x, /*xoff=*/4,
      ebn_g, ebn_b, ebn_m, ebn_v, cat, /*cat_off=*/384);

  inter_kernel<<<BATCH / 16, 256, 0, stream>>>(
      topoA, WtI, inter_wb, cat,
      ibn_g, ibn_b, ibn_m, ibn_v,
      hk, hbn_g, hbn_b, hbn_m, hbn_v, ok, ob, (float*)d_out);
}
